// TransformerBlock_78786880078280
// MI455X (gfx1250) — matrix operand order repaired and checked by reading, compiled, not run on hardware
//
#include <hip/hip_runtime.h>
#include <hip/hip_bf16.h>

#define HIDDEN   2048
#define HEADS    16
#define HEAD_DIM 128
#define INTER    8192
#define SEQ      2048
#define BATCH    2
#define NTOK     (BATCH * SEQ)   // 4096
#define EPS_RMS  1e-6f

typedef __bf16 v16bf __attribute__((ext_vector_type(16)));
typedef __bf16 v8bf  __attribute__((ext_vector_type(8)));
typedef __bf16 v4bf  __attribute__((ext_vector_type(4)));
typedef float  v8f   __attribute__((ext_vector_type(8)));

__device__ inline v16bf cat8(v8bf lo, v8bf hi) {
  return __builtin_shufflevector(lo, hi, 0,1,2,3,4,5,6,7,8,9,10,11,12,13,14,15);
}
__device__ inline v8f wmma_bf16(v16bf a, v16bf b, v8f c) {
  return __builtin_amdgcn_wmma_f32_16x16x32_bf16(false, a, false, b, (short)0, c, false, false);
}
__device__ inline v8f v8f_zero() { v8f z; for (int i = 0; i < 8; ++i) z[i] = 0.f; return z; }

// gfx1250 async copy global -> LDS (ASYNCcnt), 16 bytes per lane.
__device__ inline void async_copy_b128(unsigned lds_off, const void* gptr) {
  asm volatile("global_load_async_to_lds_b128 %0, %1, off"
               :: "v"(lds_off), "v"((unsigned long long)(uintptr_t)gptr)
               : "memory");
}
__device__ inline unsigned lds_off_of(const void* p) {
  return (unsigned)(uintptr_t)p;   // flat LDS address: offset lives in addr[31:0]
}

// ---------------------------------------------------------------- cast f32->bf16
__global__ void cast_f32_to_bf16(const float* __restrict__ s, __bf16* __restrict__ d, size_t n4) {
  size_t i = (size_t)blockIdx.x * blockDim.x + threadIdx.x;
  size_t stride = (size_t)gridDim.x * blockDim.x;
  for (size_t idx = i; idx < n4; idx += stride) {
    float4 v = ((const float4*)s)[idx];
    v4bf o;
    o[0] = (__bf16)v.x; o[1] = (__bf16)v.y; o[2] = (__bf16)v.z; o[3] = (__bf16)v.w;
    ((v4bf*)d)[idx] = o;
  }
}

// ---------------------------------------------------------------- RMSNorm -> bf16
__global__ __launch_bounds__(256)
void rmsnorm_to_bf16(const float* __restrict__ x, const float* __restrict__ w,
                     __bf16* __restrict__ out) {
  int row = blockIdx.x;
  const float* xr = x + (size_t)row * HIDDEN;
  float ss = 0.f;
  for (int i = threadIdx.x; i < HIDDEN; i += 256) { float v = xr[i]; ss += v * v; }
  for (int m = 16; m >= 1; m >>= 1) ss += __shfl_xor(ss, m, 32);
  __shared__ float red[8];
  __shared__ float snorm;
  if ((threadIdx.x & 31) == 0) red[threadIdx.x >> 5] = ss;
  __syncthreads();
  if (threadIdx.x == 0) {
    float t = 0.f;
    for (int i = 0; i < 8; ++i) t += red[i];
    snorm = rsqrtf(t * (1.0f / HIDDEN) + EPS_RMS);
  }
  __syncthreads();
  float sc = snorm;
  __bf16* orow = out + (size_t)row * HIDDEN;
  for (int i = threadIdx.x; i < HIDDEN; i += 256) orow[i] = (__bf16)(xr[i] * sc * w[i]);
}

// ---------------------------------------------------------------- GEMM: C[M,N] = A[M,K] * W[N,K]^T
// modes: 0: Cf=acc   1: Cf=acc+resid   2: Cb=bf16(acc)   3: Cb=bf16(silu(gatef)*acc)
#define BM 128
#define BN 128
#define BK 32
#define LPAD 40

__global__ __launch_bounds__(256)
void gemm_bf16_wmma(const __bf16* __restrict__ A, const __bf16* __restrict__ W,
                    int M, int N, int K,
                    float* __restrict__ Cf, __bf16* __restrict__ Cb,
                    const float* __restrict__ resid, const float* __restrict__ gatef,
                    int mode) {
  __shared__ __bf16 As[2][BM][LPAD];
  __shared__ __bf16 Ws[2][BN][LPAD];
  const int tid   = threadIdx.x;
  const int lane  = tid & 31;
  const int wave  = tid >> 5;      // 0..7
  const int waveM = wave >> 1;     // 0..3  (32 rows each)
  const int waveN = wave & 1;      // 0..1  (64 cols each)
  const int m16   = lane & 15;
  const int half  = lane >> 4;

  const int blockM = blockIdx.y * BM;
  const int blockN = blockIdx.x * BN;

  v8f acc[2][4];
  for (int i = 0; i < 2; ++i)
    for (int j = 0; j < 4; ++j) acc[i][j] = v8f_zero();

  const int ldRow = tid >> 1;           // 0..127
  const int ldCol = (tid & 1) * 16;     // 0 / 16

  const __bf16* aRow = A + (size_t)(blockM + ldRow) * K + ldCol;
  const __bf16* wRow = W + (size_t)(blockN + ldRow) * K + ldCol;

  const int nIter = K / BK;

  // prologue: tile 0 -> buffer 0
  async_copy_b128(lds_off_of(&As[0][ldRow][ldCol]),     aRow);
  async_copy_b128(lds_off_of(&As[0][ldRow][ldCol + 8]), aRow + 8);
  async_copy_b128(lds_off_of(&Ws[0][ldRow][ldCol]),     wRow);
  async_copy_b128(lds_off_of(&Ws[0][ldRow][ldCol + 8]), wRow + 8);

  for (int it = 0; it < nIter; ++it) {
    const int buf = it & 1;
    if (it + 1 < nIter) {
      const int nb = buf ^ 1;
      const __bf16* ag = aRow + (size_t)(it + 1) * BK;
      const __bf16* wg = wRow + (size_t)(it + 1) * BK;
      async_copy_b128(lds_off_of(&As[nb][ldRow][ldCol]),     ag);
      async_copy_b128(lds_off_of(&As[nb][ldRow][ldCol + 8]), ag + 8);
      async_copy_b128(lds_off_of(&Ws[nb][ldRow][ldCol]),     wg);
      async_copy_b128(lds_off_of(&Ws[nb][ldRow][ldCol + 8]), wg + 8);
      asm volatile("s_wait_asynccnt 0x4" ::: "memory");   // oldest 4 (current buf) done
    } else {
      asm volatile("s_wait_asynccnt 0x0" ::: "memory");
    }
    __syncthreads();

    v16bf afrag[2];
#pragma unroll
    for (int ms = 0; ms < 2; ++ms) {
      const __bf16* p = &As[buf][waveM * 32 + ms * 16 + m16][0];
      afrag[ms] = cat8(*(const v8bf*)(p + half * 16), *(const v8bf*)(p + half * 16 + 8));
    }
#pragma unroll
    for (int ns = 0; ns < 4; ++ns) {
      const __bf16* p = &Ws[buf][waveN * 64 + ns * 16 + m16][half * 16];
      v16bf bfrag = cat8(*(const v8bf*)p, *(const v8bf*)(p + 8));
#pragma unroll
      for (int ms = 0; ms < 2; ++ms)
        acc[ms][ns] = wmma_bf16(afrag[ms], bfrag, acc[ms][ns]);
    }
    __syncthreads();   // everyone done reading buf before it is refilled at it+2
  }

#pragma unroll
  for (int ms = 0; ms < 2; ++ms)
#pragma unroll
    for (int ns = 0; ns < 4; ++ns)
#pragma unroll
      for (int r = 0; r < 8; ++r) {
        int gr = blockM + waveM * 32 + ms * 16 + r + 8 * half;
        int gc = blockN + waveN * 64 + ns * 16 + m16;
        size_t idx = (size_t)gr * N + gc;
        float v = acc[ms][ns][r];
        if (mode == 0)       Cf[idx] = v;
        else if (mode == 1)  Cf[idx] = v + resid[idx];
        else if (mode == 2)  Cb[idx] = (__bf16)v;
        else {
          float g = gatef[idx];
          float s = g / (1.f + __expf(-g));
          Cb[idx] = (__bf16)(s * v);
        }
      }
}

// ---------------------------------------------------------------- RoPE (in place, bf16)
__global__ __launch_bounds__(256)
void rope_bf16(__bf16* __restrict__ X) {
  int idx = blockIdx.x * blockDim.x + threadIdx.x;   // over NTOK*HEADS*64
  int d    = idx & 63;
  int t    = idx >> 6;
  int head = t & (HEADS - 1);
  int tok  = t >> 4;
  int pos  = tok & (SEQ - 1);
  float freq = __powf(10000.f, -(float)(2 * d) * (1.0f / 128.0f));
  float ang = (float)pos * freq;
  float sn, cs;
  __sincosf(ang, &sn, &cs);
  size_t base = (size_t)tok * HIDDEN + head * HEAD_DIM;
  float x1 = (float)X[base + d];
  float x2 = (float)X[base + d + 64];
  X[base + d]      = (__bf16)(x1 * cs - x2 * sn);
  X[base + d + 64] = (__bf16)(x2 * cs + x1 * sn);
}

// ---------------------------------------------------------------- flash attention
// 8 waves per workgroup; 128 query rows share each staged K/V block (8x less
// K/V read traffic than 1 wave per block). Each wave owns 16 q rows and a
// private Ps staging tile. K staged via async global->LDS; V transposed manually.
#define FA_WAVES 8
#define FA_QTILE (FA_WAVES * 16)   // 128

__global__ __launch_bounds__(256)
void flash_attn_bf16(const __bf16* __restrict__ Q, const __bf16* __restrict__ Kg,
                     const __bf16* __restrict__ Vg, __bf16* __restrict__ O) {
  __shared__ __bf16 Ks[32][136];             // [key][dim], padded
  __shared__ __bf16 Vt[HEAD_DIM][40];        // [dim][key], padded
  __shared__ __bf16 Ps[FA_WAVES][16][40];    // per-wave [qrow][key], padded
  const int tid  = threadIdx.x;
  const int lane = tid & 31;
  const int wave = tid >> 5;
  const int m16  = lane & 15;
  const int half = lane >> 4;
  const int bh = blockIdx.y;
  const int b  = bh / HEADS, hh = bh % HEADS;
  const int qBase = blockIdx.x * FA_QTILE;
  const int qWave = qBase + wave * 16;
  const size_t tokBase = (size_t)b * SEQ;
  const float scale = 0.08838834764831845f;   // 1/sqrt(128)

  const __bf16* qp = Q + (tokBase + qWave + m16) * HIDDEN + hh * HEAD_DIM;
  v16bf qfrag[4];
#pragma unroll
  for (int kc = 0; kc < 4; ++kc)
    qfrag[kc] = cat8(*(const v8bf*)(qp + kc * 32 + half * 16),
                     *(const v8bf*)(qp + kc * 32 + half * 16 + 8));

  v8f acc[8];
  for (int n = 0; n < 8; ++n) acc[n] = v8f_zero();
  float mrow[8], lrow[8];
  for (int r = 0; r < 8; ++r) { mrow[r] = -1e30f; lrow[r] = 0.f; }

  const int ldRow = tid >> 3;          // 0..31 (key within block)
  const int ldCol = (tid & 7) * 16;    // 0..112 (dim)

  for (int kb = 0; kb <= qBase + FA_QTILE - 1; kb += 32) {
    // stage K block [32 x 128] via async copies; V block transposed [128 x 32]
    const __bf16* krow = Kg + (tokBase + kb + ldRow) * HIDDEN + hh * HEAD_DIM + ldCol;
    async_copy_b128(lds_off_of(&Ks[ldRow][ldCol]),     krow);
    async_copy_b128(lds_off_of(&Ks[ldRow][ldCol + 8]), krow + 8);
    const __bf16* vrow = Vg + (tokBase + kb + ldRow) * HIDDEN + hh * HEAD_DIM + ldCol;
    v8bf va = *(const v8bf*)vrow;
    v8bf vb = *(const v8bf*)(vrow + 8);
#pragma unroll
    for (int e = 0; e < 8; ++e) Vt[ldCol + e][ldRow] = va[e];
#pragma unroll
    for (int e = 0; e < 8; ++e) Vt[ldCol + 8 + e][ldRow] = vb[e];
    asm volatile("s_wait_asynccnt 0x0" ::: "memory");
    __syncthreads();

    if (kb <= qWave + 15) {   // wave-uniform causal guard (EXEC stays all-ones)
      v8f s0 = v8f_zero(), s1 = v8f_zero();
#pragma unroll
      for (int kc = 0; kc < 4; ++kc) {
        const __bf16* p0 = &Ks[m16][kc * 32 + half * 16];
        const __bf16* p1 = &Ks[16 + m16][kc * 32 + half * 16];
        v16bf b0 = cat8(*(const v8bf*)p0, *(const v8bf*)(p0 + 8));
        v16bf b1 = cat8(*(const v8bf*)p1, *(const v8bf*)(p1 + 8));
        s0 = wmma_bf16(qfrag[kc], b0, s0);
        s1 = wmma_bf16(qfrag[kc], b1, s1);
      }

#pragma unroll
      for (int r = 0; r < 8; ++r) {
        int row = qWave + r + 8 * half;
        int c0 = kb + m16, c1 = c0 + 16;
        float v0 = (c0 <= row) ? s0[r] * scale : -1e30f;
        float v1 = (c1 <= row) ? s1[r] * scale : -1e30f;
        float mx = fmaxf(v0, v1);
        for (int msk = 8; msk >= 1; msk >>= 1) mx = fmaxf(mx, __shfl_xor(mx, msk, 32));
        float mn = fmaxf(mrow[r], mx);
        float corr = __expf(mrow[r] - mn);
        float p0 = __expf(v0 - mn);
        float p1 = __expf(v1 - mn);
        float ps = p0 + p1;
        for (int msk = 8; msk >= 1; msk >>= 1) ps += __shfl_xor(ps, msk, 32);
        lrow[r] = lrow[r] * corr + ps;
        mrow[r] = mn;
#pragma unroll
        for (int n = 0; n < 8; ++n) acc[n][r] = acc[n][r] * corr;
        Ps[wave][r + 8 * half][m16]      = (__bf16)p0;
        Ps[wave][r + 8 * half][16 + m16] = (__bf16)p1;
      }
      // same-wave LDS ops are in-order; no barrier needed for private Ps tile
      const __bf16* pp = &Ps[wave][m16][0];
      v16bf pfrag = cat8(*(const v8bf*)(pp + half * 16), *(const v8bf*)(pp + half * 16 + 8));
#pragma unroll
      for (int n = 0; n < 8; ++n) {
        const __bf16* vp = &Vt[n * 16 + m16][half * 16];
        v16bf vfrag = cat8(*(const v8bf*)vp, *(const v8bf*)(vp + 8));
        acc[n] = wmma_bf16(pfrag, vfrag, acc[n]);
      }
    }
    __syncthreads();   // all waves done with Ks/Vt before restaging
  }

#pragma unroll
  for (int n = 0; n < 8; ++n)
#pragma unroll
    for (int r = 0; r < 8; ++r) {
      size_t row = tokBase + qWave + r + 8 * half;
      O[row * HIDDEN + hh * HEAD_DIM + n * 16 + m16] = (__bf16)(acc[n][r] / lrow[r]);
    }
}

// ---------------------------------------------------------------- launch
extern "C" void kernel_launch(void* const* d_in, const int* in_sizes, int n_in,
                              void* d_out, int out_size, void* d_ws, size_t ws_size,
                              hipStream_t stream) {
  (void)in_sizes; (void)n_in; (void)out_size; (void)ws_size;
  const float* x   = (const float*)d_in[0];
  const float* anw = (const float*)d_in[1];
  const float* qw  = (const float*)d_in[2];
  const float* kw  = (const float*)d_in[3];
  const float* vw  = (const float*)d_in[4];
  const float* ow  = (const float*)d_in[5];
  const float* fnw = (const float*)d_in[6];
  const float* gw  = (const float*)d_in[7];
  const float* uw  = (const float*)d_in[8];
  const float* dw  = (const float*)d_in[9];
  float* out = (float*)d_out;

  const size_t WH = (size_t)HIDDEN * HIDDEN;   // 4M
  const size_t WI = (size_t)INTER * HIDDEN;    // 16M
  const size_t TH = (size_t)NTOK * HIDDEN;     // 8M
  const size_t TI = (size_t)NTOK * INTER;      // 32M

  char* ws = (char*)d_ws;
  size_t off = 0;
  auto take = [&](size_t bytes) -> char* {
    char* p = ws + off;
    off += (bytes + 255) & ~(size_t)255;
    return p;
  };
  __bf16* wqb = (__bf16*)take(WH * 2);
  __bf16* wkb = (__bf16*)take(WH * 2);
  __bf16* wvb = (__bf16*)take(WH * 2);
  __bf16* wob = (__bf16*)take(WH * 2);
  __bf16* wgb = (__bf16*)take(WI * 2);
  __bf16* wub = (__bf16*)take(WI * 2);
  __bf16* wdb = (__bf16*)take(WI * 2);
  __bf16* h1   = (__bf16*)take(TH * 2);
  __bf16* qb   = (__bf16*)take(TH * 2);
  __bf16* kb   = (__bf16*)take(TH * 2);
  __bf16* vb   = (__bf16*)take(TH * 2);
  __bf16* attb = (__bf16*)take(TH * 2);
  __bf16* h2   = (__bf16*)take(TH * 2);
  __bf16* hmid = (__bf16*)take(TI * 2);
  float*  x1   = (float*)take(TH * 4);
  float*  gate = (float*)take(TI * 4);

  // 1) weights -> bf16
  cast_f32_to_bf16<<<4096, 256, 0, stream>>>(qw, wqb, WH / 4);
  cast_f32_to_bf16<<<4096, 256, 0, stream>>>(kw, wkb, WH / 4);
  cast_f32_to_bf16<<<4096, 256, 0, stream>>>(vw, wvb, WH / 4);
  cast_f32_to_bf16<<<4096, 256, 0, stream>>>(ow, wob, WH / 4);
  cast_f32_to_bf16<<<8192, 256, 0, stream>>>(gw, wgb, WI / 4);
  cast_f32_to_bf16<<<8192, 256, 0, stream>>>(uw, wub, WI / 4);
  cast_f32_to_bf16<<<8192, 256, 0, stream>>>(dw, wdb, WI / 4);

  // 2) attn RMSNorm
  rmsnorm_to_bf16<<<NTOK, 256, 0, stream>>>(x, anw, h1);

  // 3) QKV projections
  dim3 gHH(HIDDEN / BN, NTOK / BM);   // (16, 32)
  gemm_bf16_wmma<<<gHH, 256, 0, stream>>>(h1, wqb, NTOK, HIDDEN, HIDDEN, nullptr, qb, nullptr, nullptr, 2);
  gemm_bf16_wmma<<<gHH, 256, 0, stream>>>(h1, wkb, NTOK, HIDDEN, HIDDEN, nullptr, kb, nullptr, nullptr, 2);
  gemm_bf16_wmma<<<gHH, 256, 0, stream>>>(h1, wvb, NTOK, HIDDEN, HIDDEN, nullptr, vb, nullptr, nullptr, 2);

  // 4) RoPE on q, k
  int ropeBlocks = NTOK * HEADS * 64 / 256;   // 16384
  rope_bf16<<<ropeBlocks, 256, 0, stream>>>(qb);
  rope_bf16<<<ropeBlocks, 256, 0, stream>>>(kb);

  // 5) flash attention (128 q rows per workgroup)
  flash_attn_bf16<<<dim3(SEQ / FA_QTILE, BATCH * HEADS), 256, 0, stream>>>(qb, kb, vb, attb);

  // 6) output projection + residual
  gemm_bf16_wmma<<<gHH, 256, 0, stream>>>(attb, wob, NTOK, HIDDEN, HIDDEN, x1, nullptr, x, nullptr, 1);

  // 7) MLP
  rmsnorm_to_bf16<<<NTOK, 256, 0, stream>>>(x1, fnw, h2);
  dim3 gHI(INTER / BN, NTOK / BM);    // (64, 32)
  gemm_bf16_wmma<<<gHI, 256, 0, stream>>>(h2, wgb, NTOK, INTER, HIDDEN, gate, nullptr, nullptr, nullptr, 0);
  gemm_bf16_wmma<<<gHI, 256, 0, stream>>>(h2, wub, NTOK, INTER, HIDDEN, nullptr, hmid, nullptr, gate, 3);
  gemm_bf16_wmma<<<gHH, 256, 0, stream>>>(hmid, wdb, NTOK, HIDDEN, INTER, out, nullptr, x1, nullptr, 1);
}
